// Matrix_Decomposition_2D_Base_30743375905318
// MI455X (gfx1250) — matrix operand order repaired and checked by reading, not compiled, not run on hardware
//
#include <hip/hip_runtime.h>
#include <hip/hip_bf16.h>

// ---------------------------------------------------------------------------
// NMF (Hamburger) matrix decomposition for MI455X / gfx1250, wave32 WMMA.
//  - All GEMMs: v_wmma_f32_16x16x32_f16, fp32 accumulation.
//  - Wave tile 32x64 (8 WMMAs per K-step, A-fragments reused across 4 B cols).
//  - Register double-buffering: next K-step fragments are loaded before the
//    current WMMAs issue, so s_wait_loadcnt does not gate the matrix ops.
//  - Split-K (8 waves/workgroup + LDS tree reduction) for the K=4096 / small-M
//    GEMMs (gram_b, gram_c, num_b) to fill the machine.
// ---------------------------------------------------------------------------

typedef __attribute__((ext_vector_type(16))) _Float16 v16h;
typedef __attribute__((ext_vector_type(8)))  _Float16 v8h;
typedef __attribute__((ext_vector_type(8)))  float    v8f;

#define BSZ   8      // batch (B*S)
#define DD    512    // C / S
#define NNP   4096   // H*W
#define RR    64
#define NMF_EPS 1e-6f
#define NMF_INVT 100.0f

struct GArgs {
  const _Float16* A;        // M x K, row-major, leading dim lda
  const _Float16* Bt;       // N x K, row-major (i.e. B^T), leading dim ldb
  float*    C;              // optional fp32 out (M x N, ldc)
  _Float16* Ch;             // optional f16 out, same layout as C
  _Float16* ChT;            // optional f16 transposed out (N x M, ldcT)
  const float* upd_old;     // mode 1: old value, layout of C
  const float* upd_num;     // mode 1: numerator, layout of C
  long sA, sB, sC;          // per-batch element strides
  int lda, ldb, ldc, ldcT;
  int M, Nn, K;
  int mode;                 // 0 = store, 1 = multiplicative update
  int nb;                   // batches
  int kwaves;               // 1 = wave-per-tile, 8 = split-K within block
};

__device__ __forceinline__ v16h loadA16x32(const _Float16* row, int k, int grp) {
  // A fragment 16x32: lane-group g holds K = k+8g..k+8g+7 and k+16+8g..k+16+8g+7
  v8h lo = *(const v8h*)(row + k + grp * 8);
  v8h hi = *(const v8h*)(row + k + 16 + grp * 8);
  v16h f;
#pragma unroll
  for (int i = 0; i < 8; ++i) { f[i] = lo[i]; f[i + 8] = hi[i]; }
  return f;
}

__device__ __forceinline__ v16h loadB32x16(const _Float16* row, int k, int grp) {
  // B fragment 32x16: lane-group g holds K = k+8g..k+8g+7 and k+16+8g..k+16+8g+7
  v8h lo = *(const v8h*)(row + k + grp * 8);
  v8h hi = *(const v8h*)(row + k + 16 + grp * 8);
  v16h f;
#pragma unroll
  for (int i = 0; i < 8; ++i) { f[i] = lo[i]; f[i + 8] = hi[i]; }
  return f;
}

#define WMMA(a, b, c) __builtin_amdgcn_wmma_f32_16x16x32_f16( \
    false, (a), false, (b), (short)0, (c), false, false)

// Each output block is 32 rows x 64 cols of C.
//  kwaves==1: 8 independent waves/block, one output block each.
//  kwaves==8: all 8 waves split K over one output block, LDS tree-reduce.
__global__ __launch_bounds__(256) void gemm_wmma(GArgs g) {
  __shared__ float red[4 * 2048];   // 32 KB reduction scratch
  const int lane = threadIdx.x & 31;
  const int wv   = threadIdx.x >> 5;
  const int kw   = g.kwaves;
  const int strips_m = g.M >> 5;
  const int strips_n = g.Nn >> 6;
  const int per_batch = strips_m * strips_n;
  const int total = g.nb * per_batch;

  int out_id, kslot;
  if (kw == 1) { out_id = blockIdx.x * 8 + wv; kslot = 0; }
  else         { out_id = blockIdx.x;          kslot = wv; }
  if (out_id >= total) return;      // wave-uniform (block-uniform when kw==8)

  const int b  = out_id / per_batch;
  const int t  = out_id - b * per_batch;
  const int tm = (t % strips_m) << 5;
  const int tn = (t / strips_m) << 6;
  const int m   = lane & 15;
  const int grp = lane >> 4;

  const int kslice = g.K / kw;      // always a multiple of 64 here
  const long k0 = (long)kslot * kslice;

  const _Float16* Arow0 = g.A + (long)b * g.sA + (long)(tm + m) * g.lda + k0;
  const _Float16* Arow1 = Arow0 + (long)16 * g.lda;
  const _Float16* Bbase = g.Bt + (long)b * g.sB + k0;
  const _Float16* Brow[4];
#pragma unroll
  for (int j = 0; j < 4; ++j)
    Brow[j] = Bbase + (long)(tn + j * 16 + m) * g.ldb;

  v8f acc[8] = {};                  // [ms*4 + j]

  // -------- software-pipelined K loop (double-buffered fragments) --------
  v16h a0 = loadA16x32(Arow0, 0, grp);
  v16h a1 = loadA16x32(Arow1, 0, grp);
  v16h bf0 = loadB32x16(Brow[0], 0, grp);
  v16h bf1 = loadB32x16(Brow[1], 0, grp);
  v16h bf2 = loadB32x16(Brow[2], 0, grp);
  v16h bf3 = loadB32x16(Brow[3], 0, grp);

  for (int k = 32; k < kslice; k += 32) {
    v16h na0 = loadA16x32(Arow0, k, grp);
    v16h na1 = loadA16x32(Arow1, k, grp);
    v16h nb0 = loadB32x16(Brow[0], k, grp);
    v16h nb1 = loadB32x16(Brow[1], k, grp);
    v16h nb2 = loadB32x16(Brow[2], k, grp);
    v16h nb3 = loadB32x16(Brow[3], k, grp);
    acc[0] = WMMA(a0, bf0, acc[0]);
    acc[1] = WMMA(a0, bf1, acc[1]);
    acc[2] = WMMA(a0, bf2, acc[2]);
    acc[3] = WMMA(a0, bf3, acc[3]);
    acc[4] = WMMA(a1, bf0, acc[4]);
    acc[5] = WMMA(a1, bf1, acc[5]);
    acc[6] = WMMA(a1, bf2, acc[6]);
    acc[7] = WMMA(a1, bf3, acc[7]);
    a0 = na0; a1 = na1;
    bf0 = nb0; bf1 = nb1; bf2 = nb2; bf3 = nb3;
  }
  acc[0] = WMMA(a0, bf0, acc[0]);
  acc[1] = WMMA(a0, bf1, acc[1]);
  acc[2] = WMMA(a0, bf2, acc[2]);
  acc[3] = WMMA(a0, bf3, acc[3]);
  acc[4] = WMMA(a1, bf0, acc[4]);
  acc[5] = WMMA(a1, bf1, acc[5]);
  acc[6] = WMMA(a1, bf2, acc[6]);
  acc[7] = WMMA(a1, bf3, acc[7]);

  // -------- cross-wave split-K reduction (LDS tree) --------
  if (kw > 1) {
#pragma unroll
    for (int step = 4; step >= 1; step >>= 1) {
      if (wv >= step && wv < 2 * step) {
        float* dst = &red[(wv - step) * 2048];
        int e = 0;
#pragma unroll
        for (int q = 0; q < 8; ++q)
#pragma unroll
          for (int i = 0; i < 8; ++i, ++e)
            dst[e * 32 + lane] = acc[q][i];
      }
      __syncthreads();
      if (wv < step) {
        const float* src = &red[wv * 2048];
        int e = 0;
#pragma unroll
        for (int q = 0; q < 8; ++q)
#pragma unroll
          for (int i = 0; i < 8; ++i, ++e)
            acc[q][i] += src[e * 32 + lane];
      }
      __syncthreads();
    }
    if (wv != 0) return;            // wave 0 holds the full sum
  }

  // -------- epilogue --------
  // C/D layout: lane holds col = lane&15; VGPR i holds row i + 8*grp.
#pragma unroll
  for (int ms = 0; ms < 2; ++ms)
#pragma unroll
    for (int j = 0; j < 4; ++j) {
      const int c = tn + j * 16 + m;
#pragma unroll
      for (int i = 0; i < 8; ++i) {
        const int r = tm + ms * 16 + i + 8 * grp;
        const long idx = (long)b * g.sC + (long)r * g.ldc + c;
        float v = acc[ms * 4 + j][i];
        if (g.mode == 1)
          v = g.upd_old[idx] * g.upd_num[idx] / (v + NMF_EPS);
        if (g.C)   g.C[idx]  = v;
        if (g.Ch)  g.Ch[idx] = (_Float16)v;
        if (g.ChT) g.ChT[(long)b * g.sC + (long)c * g.ldcT + r] = (_Float16)v;
      }
    }
}

// x (BS, D, N) fp32 -> f16 row-major and f16 transposed (N, D)
__global__ __launch_bounds__(256) void prep_x(const float* __restrict__ x,
                                              _Float16* __restrict__ xh,
                                              _Float16* __restrict__ xhT) {
  long i = (long)blockIdx.x * 256 + threadIdx.x;
  if (i >= (long)BSZ * DD * NNP) return;
  int n = (int)(i % NNP);
  long t = i / NNP;
  int d = (int)(t % DD);
  int b = (int)(t / DD);
  _Float16 h = (_Float16)x[i];
  xh[i] = h;
  xhT[(long)b * DD * NNP + (long)n * DD + d] = h;
}

// bases (BS, D, R) fp32 -> fp32 working copy + f16 + f16 transposed (R, D)
__global__ __launch_bounds__(256) void prep_bases(const float* __restrict__ bin,
                                                  float* __restrict__ bF,
                                                  _Float16* __restrict__ bH,
                                                  _Float16* __restrict__ bTH) {
  int i = blockIdx.x * 256 + threadIdx.x;
  if (i >= BSZ * DD * RR) return;
  int r = i % RR;
  int t = i / RR;
  int d = t % DD;
  int b = t / DD;
  float v = bin[i];
  bF[i] = v;
  bH[i] = (_Float16)v;
  bTH[(b * RR + r) * DD + d] = (_Float16)v;
}

// coef = softmax(INV_T * num, axis=r); one wave per row (R=64 -> 2 vals/lane)
__global__ __launch_bounds__(256) void softmax_init(const float* __restrict__ num,
                                                    float* __restrict__ coefF,
                                                    _Float16* __restrict__ coefH,
                                                    _Float16* __restrict__ coefTH) {
  const int lane = threadIdx.x & 31;
  const int row  = blockIdx.x * 8 + (threadIdx.x >> 5);
  if (row >= BSZ * NNP) return;
  const int b = row / NNP;
  const int n = row - b * NNP;
  const float* p = num + (long)row * RR;
  float v0 = NMF_INVT * p[lane];
  float v1 = NMF_INVT * p[lane + 32];
  float mx = fmaxf(v0, v1);
  for (int off = 16; off; off >>= 1) mx = fmaxf(mx, __shfl_xor(mx, off, 32));
  float e0 = __expf(v0 - mx);
  float e1 = __expf(v1 - mx);
  float s = e0 + e1;
  for (int off = 16; off; off >>= 1) s += __shfl_xor(s, off, 32);
  float inv = 1.0f / s;
  e0 *= inv; e1 *= inv;
  const long base = (long)row * RR;
  coefF[base + lane]      = e0;
  coefF[base + lane + 32] = e1;
  coefH[base + lane]      = (_Float16)e0;
  coefH[base + lane + 32] = (_Float16)e1;
  const long tb = (long)b * NNP * RR;
  coefTH[tb + (long)lane * NNP + n]        = (_Float16)e0;
  coefTH[tb + (long)(lane + 32) * NNP + n] = (_Float16)e1;
}

static inline void launch_gemm(const GArgs& g, hipStream_t s) {
  int total  = g.nb * (g.M >> 5) * (g.Nn >> 6);
  int blocks = (g.kwaves == 1) ? (total + 7) / 8 : total;
  hipLaunchKernelGGL(gemm_wmma, dim3(blocks), dim3(256), 0, s, g);
}

extern "C" void kernel_launch(void* const* d_in, const int* in_sizes, int n_in,
                              void* d_out, int out_size, void* d_ws, size_t ws_size,
                              hipStream_t stream) {
  (void)in_sizes; (void)n_in; (void)out_size; (void)ws_size;
  const float* x_in     = (const float*)d_in[0];
  const float* bases_in = (const float*)d_in[1];
  float* out = (float*)d_out;

  // ---- carve workspace (~95 MB) ----
  char* p = (char*)d_ws;
  auto carve = [&](size_t bytes) -> char* {
    char* r = p; p += (bytes + 255) & ~(size_t)255; return r;
  };
  _Float16* xh      = (_Float16*)carve((size_t)BSZ * DD * NNP * 2);
  _Float16* xhT     = (_Float16*)carve((size_t)BSZ * DD * NNP * 2);
  float*    basesF  = (float*)   carve((size_t)BSZ * DD * RR * 4);
  _Float16* basesH  = (_Float16*)carve((size_t)BSZ * DD * RR * 2);
  _Float16* basesTH = (_Float16*)carve((size_t)BSZ * DD * RR * 2);
  float*    coefF   = (float*)   carve((size_t)BSZ * NNP * RR * 4);
  _Float16* coefH   = (_Float16*)carve((size_t)BSZ * NNP * RR * 2);
  _Float16* coefTH  = (_Float16*)carve((size_t)BSZ * NNP * RR * 2);
  float*    numF    = (float*)   carve((size_t)BSZ * NNP * RR * 4);
  float*    numbF   = (float*)   carve((size_t)BSZ * DD * RR * 4);
  _Float16* gramBH  = (_Float16*)carve((size_t)BSZ * RR * RR * 2);
  _Float16* gramCH  = (_Float16*)carve((size_t)BSZ * RR * RR * 2);

  // ---- prep: fp32 -> f16 (+ transposed copies) ----
  {
    long total = (long)BSZ * DD * NNP;
    hipLaunchKernelGGL(prep_x, dim3((unsigned)((total + 255) / 256)), dim3(256), 0, stream,
                       x_in, xh, xhT);
    int tb = BSZ * DD * RR;
    hipLaunchKernelGGL(prep_bases, dim3((tb + 255) / 256), dim3(256), 0, stream,
                       bases_in, basesF, basesH, basesTH);
  }

  // ---- GEMM argument templates ----
  GArgs a{};  // num (N x R) = x^T . bases        K = D
  a.A = xhT; a.lda = DD;  a.sA = (long)DD * NNP;
  a.Bt = basesTH; a.ldb = DD; a.sB = (long)RR * DD;
  a.C = numF; a.ldc = RR; a.sC = (long)NNP * RR;
  a.M = NNP; a.Nn = RR; a.K = DD; a.mode = 0; a.nb = BSZ; a.kwaves = 1;

  GArgs gb{}; // gram_b (R x R) = bases^T . bases  K = D   (f16 out only)
  gb.A = basesTH; gb.lda = DD; gb.sA = (long)RR * DD;
  gb.Bt = basesTH; gb.ldb = DD; gb.sB = (long)RR * DD;
  gb.Ch = gramBH; gb.ldc = RR; gb.sC = (long)RR * RR;
  gb.M = RR; gb.Nn = RR; gb.K = DD; gb.mode = 0; gb.nb = BSZ; gb.kwaves = 8;

  GArgs cu{}; // coef update: acc = coef.gram_b; coef *= num/(acc+eps)
  cu.A = coefH; cu.lda = RR; cu.sA = (long)NNP * RR;
  cu.Bt = gramBH; cu.ldb = RR; cu.sB = (long)RR * RR;
  cu.C = coefF; cu.Ch = coefH; cu.ChT = coefTH; cu.ldcT = NNP;
  cu.upd_old = coefF; cu.upd_num = numF;
  cu.ldc = RR; cu.sC = (long)NNP * RR;
  cu.M = NNP; cu.Nn = RR; cu.K = RR; cu.mode = 1; cu.nb = BSZ; cu.kwaves = 1;

  GArgs nb{}; // num_b (D x R) = x . coef          K = N   (split-K)
  nb.A = xh; nb.lda = NNP; nb.sA = (long)DD * NNP;
  nb.Bt = coefTH; nb.ldb = NNP; nb.sB = (long)NNP * RR;
  nb.C = numbF; nb.ldc = RR; nb.sC = (long)DD * RR;
  nb.M = DD; nb.Nn = RR; nb.K = NNP; nb.mode = 0; nb.nb = BSZ; nb.kwaves = 8;

  GArgs gc{}; // gram_c (R x R) = coef^T . coef    K = N   (split-K, f16 out)
  gc.A = coefTH; gc.lda = NNP; gc.sA = (long)NNP * RR;
  gc.Bt = coefTH; gc.ldb = NNP; gc.sB = (long)NNP * RR;
  gc.Ch = gramCH; gc.ldc = RR; gc.sC = (long)RR * RR;
  gc.M = RR; gc.Nn = RR; gc.K = NNP; gc.mode = 0; gc.nb = BSZ; gc.kwaves = 8;

  GArgs bu{}; // bases update: acc = bases.gram_c; bases *= num_b/(acc+eps)
  bu.A = basesH; bu.lda = RR; bu.sA = (long)DD * RR;
  bu.Bt = gramCH; bu.ldb = RR; bu.sB = (long)RR * RR;
  bu.C = basesF; bu.Ch = basesH; bu.ChT = basesTH; bu.ldcT = DD;
  bu.upd_old = basesF; bu.upd_num = numbF;
  bu.ldc = RR; bu.sC = (long)DD * RR;
  bu.M = DD; bu.Nn = RR; bu.K = RR; bu.mode = 1; bu.nb = BSZ; bu.kwaves = 1;

  GArgs oo{}; // out (D x N) = bases . coef^T      K = R
  oo.A = basesH; oo.lda = RR; oo.sA = (long)DD * RR;
  oo.Bt = coefH; oo.ldb = RR; oo.sB = (long)NNP * RR;
  oo.C = out; oo.ldc = NNP; oo.sC = (long)DD * NNP;
  oo.M = DD; oo.Nn = NNP; oo.K = RR; oo.mode = 0; oo.nb = BSZ; oo.kwaves = 1;

  // ---- pipeline ----
  launch_gemm(a, stream);            // num0 (initial bases)
  launch_gemm(gb, stream);           // gram_b (initial bases)
  hipLaunchKernelGGL(softmax_init, dim3((BSZ * NNP + 7) / 8), dim3(256), 0, stream,
                     numF, coefF, coefH, coefTH);

  for (int it = 0; it < 7; ++it) {
    if (it) {                        // it==0 reuses num0/gram_b (same bases)
      launch_gemm(a, stream);
      launch_gemm(gb, stream);
    }
    launch_gemm(cu, stream);         // coef update
    launch_gemm(nb, stream);         // num_b
    launch_gemm(gc, stream);         // gram_c
    launch_gemm(bu, stream);         // bases update
  }

  // compute_coef: one extra coef update, then reconstruction
  launch_gemm(a, stream);
  launch_gemm(gb, stream);
  launch_gemm(cu, stream);
  launch_gemm(oo, stream);
}
